// MultiHeadAttention_76338748719257
// MI455X (gfx1250) — matrix operand order repaired and checked by reading, not compiled, not run on hardware
//
#include <hip/hip_runtime.h>

// ---------------- problem constants ----------------
#define E_   1024
#define H_   16
#define D_   64
#define S_   2048
#define B_   2
#define BSE_ (B_ * S_ * E_)
#define SCALE_ 0.03125f   // 1/sqrt(1024)

typedef __attribute__((ext_vector_type(16))) _Float16 v16h;
typedef __attribute__((ext_vector_type(8)))  _Float16 h8;
typedef __attribute__((ext_vector_type(8)))  float    v8f;
typedef __attribute__((ext_vector_type(4)))  float    f4;

typedef __attribute__((ext_vector_type(4))) unsigned int u32x4;
typedef __attribute__((ext_vector_type(8))) int          i32x8;
typedef __attribute__((ext_vector_type(4))) int          i32x4;

#if __has_builtin(__builtin_amdgcn_tensor_load_to_lds) && \
    __has_builtin(__builtin_amdgcn_s_wait_tensorcnt)
#define USE_TDM 1
#else
#define USE_TDM 0
#endif

static __device__ __forceinline__ v16h cat16(h8 lo, h8 hi) {
    return __builtin_shufflevector(lo, hi, 0,1,2,3,4,5,6,7,8,9,10,11,12,13,14,15);
}

// A-fragment loaders: 16 halfs/lane, contiguous k order (same order as the B loaders)
// half i -> K = k0 + 16g + i
static __device__ __forceinline__ v16h load_a_frag(const float* ap, int g) {
    f4 x0 = *(const f4*)(ap + 16 * g);
    f4 x1 = *(const f4*)(ap + 16 * g + 4);
    f4 x2 = *(const f4*)(ap + 16 * g + 8);
    f4 x3 = *(const f4*)(ap + 16 * g + 12);
    v16h a;
#pragma unroll
    for (int i = 0; i < 4; ++i) {
        a[i]      = (_Float16)x0[i];
        a[4 + i]  = (_Float16)x1[i];
        a[8 + i]  = (_Float16)x2[i];
        a[12 + i] = (_Float16)x3[i];
    }
    return a;
}
static __device__ __forceinline__ v16h load_a_frag(const _Float16* ap, int g) {
    return cat16(*(const h8*)(ap + 16 * g), *(const h8*)(ap + 16 * g + 8));
}

// B-fragment from 16 contiguous f32 (K-major): lane holds col n=lane&15, K = k0+16g+i
static __device__ __forceinline__ v16h load_b_frag_f32(const float* wp) {
    f4 w0 = *(const f4*)(wp);
    f4 w1 = *(const f4*)(wp + 4);
    f4 w2 = *(const f4*)(wp + 8);
    f4 w3 = *(const f4*)(wp + 12);
    v16h b;
#pragma unroll
    for (int i = 0; i < 4; ++i) {
        b[i]      = (_Float16)w0[i];
        b[4 + i]  = (_Float16)w1[i];
        b[8 + i]  = (_Float16)w2[i];
        b[12 + i] = (_Float16)w3[i];
    }
    return b;
}

// ---------------------------------------------------------------------------
// GEMM:  Out[m, n] = sum_k X[m, k] * W[n, k] + bias[n]        (x @ W^T + b)
// Block tile 64(M) x 128(N), 256 threads = 8 waves, wave tile 16 x 64.
// ---------------------------------------------------------------------------
template <typename TIn, typename TOut>
__global__ __launch_bounds__(256)
void mha_proj_gemm(const TIn* __restrict__ X, const float* __restrict__ W,
                   const float* __restrict__ bias, TOut* __restrict__ Out) {
    const int lane = threadIdx.x & 31;
    const int wid  = threadIdx.x >> 5;
    const int g    = lane >> 4;
    const int l16  = lane & 15;
    const int wm   = wid & 3;
    const int wn   = wid >> 2;

    const int m_base = blockIdx.x * 64 + 16 * wm;
    const int n_base = blockIdx.y * 128 + 64 * wn;
    const int row    = m_base + l16;

    v8f acc[4] = {};

    for (int k0 = 0; k0 < E_; k0 += 32) {
        const v16h a = load_a_frag(X + (size_t)row * E_ + k0, g);
#pragma unroll
        for (int j = 0; j < 4; ++j) {
            const int n  = n_base + 16 * j + l16;
            const v16h b = load_b_frag_f32(W + (size_t)n * E_ + k0 + 16 * g);
            acc[j] = __builtin_amdgcn_wmma_f32_16x16x32_f16(
                false, a, false, b, (short)0, acc[j], false, false);
        }
    }

#pragma unroll
    for (int j = 0; j < 4; ++j) {
        const int   n  = n_base + 16 * j + l16;
        const float bs = bias[n];
#pragma unroll
        for (int r = 0; r < 8; ++r) {
            const int m = m_base + r + 8 * g;   // C layout: M = r + 8*(lane/16)
            Out[(size_t)m * E_ + n] = (TOut)(acc[j][r] + bs);
        }
    }
}

// ---------------------------------------------------------------------------
// Flash attention over the raw [B,H,S,D] view.
// One block = one (b,h) and 128 query rows. 8 waves x 16 query rows each.
// Key loop in tiles of 32.  K tile is DMA'd into LDS by the Tensor Data
// Mover (wave 0 issues tensor_load_to_lds, TENSORcnt-tracked, descriptor
// padding produces the 72-half LDS row stride); V tile is staged transposed
// by all 256 threads, with prefetch of the next tile.
// ---------------------------------------------------------------------------
#define KSTR 72   // LDS stride (halfs) for K tile  [32][KSTR]
#define VSTR 40   // LDS stride (halfs) for V^T tile [64][VSTR]
#define PSTR 40   // LDS stride (halfs) for per-wave P tile [16][PSTR]

__global__ __launch_bounds__(256)
void mha_flash_attn(const _Float16* __restrict__ Qp, const _Float16* __restrict__ Kp,
                    const _Float16* __restrict__ Vp, _Float16* __restrict__ Op) {
    __shared__ __align__(16) _Float16 sK[32 * KSTR];
    __shared__ __align__(16) _Float16 sV[64 * VSTR];
    __shared__ __align__(16) _Float16 sP[8 * 16 * PSTR];

    const int lane = threadIdx.x & 31;
    const int wid  = threadIdx.x >> 5;
    const int g    = lane >> 4;
    const int l16  = lane & 15;

    const int qblk = blockIdx.x & 15;   // 16 query chunks of 128
    const int bh   = blockIdx.x >> 4;   // 0..31 = b*H + h

    const _Float16* Qh = Qp + (size_t)bh * S_ * D_;
    const _Float16* Kh = Kp + (size_t)bh * S_ * D_;
    const _Float16* Vh = Vp + (size_t)bh * S_ * D_;
    _Float16*       Oh = Op + (size_t)bh * S_ * D_;

    const int q0   = qblk * 128;
    const int qrow = q0 + 16 * wid + l16;

    // Preload Q fragments (2 chunks of K=32 over D=64), fold in softmax scale.
    v16h qf[2];
#pragma unroll
    for (int c = 0; c < 2; ++c) {
        v16h t = load_a_frag(Qh + (size_t)qrow * D_ + 32 * c, g);
#pragma unroll
        for (int i = 0; i < 16; ++i) t[i] = t[i] * (_Float16)SCALE_;
        qf[c] = t;
    }

    v8f   o[4] = {};
    float mrun[8], lrun[8];
#pragma unroll
    for (int r = 0; r < 8; ++r) { mrun[r] = -__builtin_inff(); lrun[r] = 0.0f; }

    _Float16* sPw = sP + wid * (16 * PSTR);
    const int ldkey = threadIdx.x >> 3;        // 0..31 : key row staged
    const int ldseg = (threadIdx.x & 7) * 8;   // 0..56 : d segment staged

#if USE_TDM
    const unsigned sK_lds = (unsigned)(unsigned long long)(const void*)sK;
#endif

    for (int kt = 0; kt < S_; kt += 32) {
        __syncthreads();

#if USE_TDM
        // --- K tile: TDM DMA global->LDS, one descriptor per block ---
        if (wid == 0) {
            const unsigned long long ga =
                (unsigned long long)(const void*)(Kh + (size_t)kt * D_);
            u32x4 g0;
            g0[0] = 1u;                                        // count=1, gather off
            g0[1] = sK_lds;                                    // lds_addr
            g0[2] = (unsigned)(ga & 0xFFFFFFFFu);              // global_addr[31:0]
            g0[3] = (unsigned)((ga >> 32) & 0x01FFFFFFu)       // global_addr[56:32]
                    | (2u << 30);                              // type=2 (image)
            i32x8 g1;
            g1[0] = (int)((1u << 16)      // data_size = 2 bytes
                        | (1u << 20)      // pad_enable
                        | (4u << 22)      // pad_interval: 32 dwords (=64 halfs)
                        | (3u << 25));    // pad_amount: 4 dwords (=8 halfs) -> KSTR 72
            g1[1] = (int)(64u << 16);     // tensor_dim0 = 64  (bits[79:48] lo16)
            g1[2] = (int)(32u << 16);     // dim0 hi16=0 | tensor_dim1 = 32
            g1[3] = (int)(64u << 16);     // dim1 hi16=0 | tile_dim0 = 64
            g1[4] = 32;                   // tile_dim1 = 32 | tile_dim2 = 0
            g1[5] = 64;                   // tensor_dim0_stride = 64 (halfs)
            g1[6] = 0;
            g1[7] = 0;
            const i32x4 gz4 = {0, 0, 0, 0};
            const i32x8 gz8 = {0, 0, 0, 0, 0, 0, 0, 0};
            // amdgpu-toolchain (clang-23) 6-arg form
            __builtin_amdgcn_tensor_load_to_lds(g0, g1, gz4, gz4, gz8, 0);
        }
#else
        {   // fallback: manual K tile staging
            h8 kv = *(const h8*)(Kh + (size_t)(kt + ldkey) * D_ + ldseg);
            *(h8*)(sK + ldkey * KSTR + ldseg) = kv;
        }
#endif

        {   // --- V tile staged transposed: sV[d][key] ---
            h8 vv = *(const h8*)(Vh + (size_t)(kt + ldkey) * D_ + ldseg);
#pragma unroll
            for (int i = 0; i < 8; ++i) sV[(ldseg + i) * VSTR + ldkey] = vv[i];
            if (kt + 32 < S_) {  // warm L2 one tile ahead (global_prefetch_b8)
                __builtin_prefetch(Vh + (size_t)(kt + 32 + ldkey) * D_ + ldseg, 0, 1);
                __builtin_prefetch(Kh + (size_t)(kt + 32 + ldkey) * D_ + ldseg, 0, 1);
            }
        }

#if USE_TDM
        if (wid == 0) __builtin_amdgcn_s_wait_tensorcnt(0);
#endif
        __syncthreads();

        // ---- scores: 16 queries x 32 keys, two 16x16 tiles ----
        v8f sc[2];
#pragma unroll
        for (int j = 0; j < 2; ++j) {
            const int key = 16 * j + l16;  // B frag: col = lane&15, K = 16g+i
            v16h kb0 = cat16(*(const h8*)(sK + key * KSTR + 16 * g),
                             *(const h8*)(sK + key * KSTR + 16 * g + 8));
            v16h kb1 = cat16(*(const h8*)(sK + key * KSTR + 32 + 16 * g),
                             *(const h8*)(sK + key * KSTR + 32 + 16 * g + 8));
            v8f s = {};
            s = __builtin_amdgcn_wmma_f32_16x16x32_f16(false, qf[0], false, kb0,
                                                       (short)0, s, false, false);
            s = __builtin_amdgcn_wmma_f32_16x16x32_f16(false, qf[1], false, kb1,
                                                       (short)0, s, false, false);
            sc[j] = s;
        }

        // ---- online softmax: per-row max / correction ----
        float corr[8];
#pragma unroll
        for (int r = 0; r < 8; ++r) {
            float t = fmaxf(sc[0][r], sc[1][r]);
#pragma unroll
            for (int msk = 1; msk < 16; msk <<= 1)
                t = fmaxf(t, __shfl_xor(t, msk, 32));
            const float mnew = fmaxf(mrun[r], t);
            corr[r] = __expf(mrun[r] - mnew);
            mrun[r] = mnew;
        }

        // ---- P = exp(s - m), stage to LDS, accumulate row sums ----
#pragma unroll
        for (int r = 0; r < 8; ++r) {
            const float p0   = __expf(sc[0][r] - mrun[r]);
            const float p1   = __expf(sc[1][r] - mrun[r]);
            const int   mloc = r + 8 * g;
            sPw[mloc * PSTR + l16]      = (_Float16)p0;
            sPw[mloc * PSTR + 16 + l16] = (_Float16)p1;
            float ps = p0 + p1;
#pragma unroll
            for (int msk = 1; msk < 16; msk <<= 1)
                ps += __shfl_xor(ps, msk, 32);
            lrun[r] = lrun[r] * corr[r] + ps;
        }

        // ---- rescale output accumulators ----
#pragma unroll
        for (int nj = 0; nj < 4; ++nj)
#pragma unroll
            for (int r = 0; r < 8; ++r) o[nj][r] *= corr[r];

        // ---- P (A layout) from LDS, then P @ V ----
        v16h pf = cat16(*(const h8*)(sPw + l16 * PSTR + 16 * g),
                        *(const h8*)(sPw + l16 * PSTR + 16 * g + 8));
#pragma unroll
        for (int nj = 0; nj < 4; ++nj) {
            const int d = 16 * nj + l16;   // B frag: col = d, K = key = 16g+i
            v16h vb = cat16(*(const h8*)(sV + d * VSTR + 16 * g),
                            *(const h8*)(sV + d * VSTR + 16 * g + 8));
            o[nj] = __builtin_amdgcn_wmma_f32_16x16x32_f16(false, pf, false, vb,
                                                           (short)0, o[nj], false, false);
        }
    }

    // ---- normalize and write out (f16, raw [B,H,S,D] view == flat [B,S,E]) ----
#pragma unroll
    for (int r = 0; r < 8; ++r) {
        const float inv = 1.0f / lrun[r];
        const int   m   = r + 8 * g;
#pragma unroll
        for (int nj = 0; nj < 4; ++nj) {
            const int d = 16 * nj + l16;
            Oh[(size_t)(q0 + 16 * wid + m) * D_ + d] = (_Float16)(o[nj][r] * inv);
        }
    }
}

// ---------------------------------------------------------------------------
extern "C" void kernel_launch(void* const* d_in, const int* in_sizes, int n_in,
                              void* d_out, int out_size, void* d_ws, size_t ws_size,
                              hipStream_t stream) {
    (void)in_sizes; (void)n_in; (void)out_size; (void)ws_size;

    const float* q  = (const float*)d_in[0];
    const float* k  = (const float*)d_in[1];
    const float* v  = (const float*)d_in[2];
    const float* wq = (const float*)d_in[3];
    const float* bq = (const float*)d_in[4];
    const float* wk = (const float*)d_in[5];
    const float* bk = (const float*)d_in[6];
    const float* wv = (const float*)d_in[7];
    const float* bv = (const float*)d_in[8];
    const float* wo = (const float*)d_in[9];
    const float* bo = (const float*)d_in[10];
    float*       out = (float*)d_out;

    _Float16* Qp = (_Float16*)d_ws;   // each buffer: B*S*E halfs = 8 MB
    _Float16* Kp = Qp + BSE_;
    _Float16* Vp = Kp + BSE_;
    _Float16* Ohb = Vp + BSE_;        // total workspace use: 32 MB

    const dim3 blk(256);
    const dim3 gproj(B_ * S_ / 64, E_ / 128);   // (64, 8)

    mha_proj_gemm<float, _Float16><<<gproj, blk, 0, stream>>>(q, wq, bq, Qp);
    mha_proj_gemm<float, _Float16><<<gproj, blk, 0, stream>>>(k, wk, bk, Kp);
    mha_proj_gemm<float, _Float16><<<gproj, blk, 0, stream>>>(v, wv, bv, Vp);

    mha_flash_attn<<<dim3(B_ * H_ * (S_ / 128)), blk, 0, stream>>>(Qp, Kp, Vp, Ohb);

    mha_proj_gemm<_Float16, float><<<gproj, blk, 0, stream>>>(Ohb, wo, bo, out);
}
